// SelfAttention_16990890623301
// MI455X (gfx1250) — matrix operand order repaired and checked by reading, not compiled, not run on hardware
//
#include <hip/hip_runtime.h>
#include <stdint.h>

#define BATCH 2
#define SEQ   4096
#define DIM   1024

typedef __attribute__((ext_vector_type(16))) __bf16 v16bf;
typedef __attribute__((ext_vector_type(8)))  float  v8f;
typedef unsigned int v4u __attribute__((ext_vector_type(4)));
typedef int          v4i __attribute__((ext_vector_type(4)));
typedef int          v8i __attribute__((ext_vector_type(8)));

struct Pack32 { uint4 lo, hi; };   // 32 bytes == one v16bf fragment

#if __has_builtin(__builtin_amdgcn_tensor_load_to_lds) && \
    __has_builtin(__builtin_amdgcn_s_wait_tensorcnt)
#define HAVE_TDM 1
#else
#define HAVE_TDM 0
#endif

__device__ __forceinline__ unsigned short f32_to_bf16(float f) {
  unsigned int u = __builtin_bit_cast(unsigned int, f);
  u += 0x7FFFu + ((u >> 16) & 1u);          // round-to-nearest-even
  return (unsigned short)(u >> 16);
}

// 16-bit A fragment (16 x 32 slice).  ISA layout: lane<16 -> M=lane, K in
// {kb..kb+7, kb+16..kb+23} with kb=0; lane>=16 -> same M, kb=8.
__device__ __forceinline__ v16bf load_frag_a(const unsigned short* base, int ld, int lane) {
  int m  = lane & 15;
  int kb = (lane < 16) ? 0 : 8;
  const unsigned short* p = base + (size_t)m * ld + kb;
  Pack32 pk;
  pk.lo = *(const uint4*)(p);        // e0..e7  = K kb..kb+7
  pk.hi = *(const uint4*)(p + 16);   // e8..e15 = K kb+16..kb+23
  return __builtin_bit_cast(v16bf, pk);
}

// 16-bit B fragment (32 x 16).  Storage is "one row per N": element(k) at
// base + n*ld + k.  lane<16 -> N=lane, K={0..7,16..23} ; lane>=16 -> K={8..15,24..31}.
__device__ __forceinline__ v16bf load_frag_b(const unsigned short* base, int ld, int lane) {
  int n  = lane & 15;
  int kb = (lane < 16) ? 0 : 8;
  const unsigned short* p = base + (size_t)n * ld + kb;
  Pack32 pk;
  pk.lo = *(const uint4*)(p);        // K kb..kb+7
  pk.hi = *(const uint4*)(p + 16);   // K kb+16..kb+23
  return __builtin_bit_cast(v16bf, pk);
}

#if HAVE_TDM
// 2-D TDM descriptor (ISA 8.3/8.4): data_size=1 (2 bytes/elem), no padding,
// no multicast, no iteration.  tile0 x tile1 elements, row stride stride0.
__device__ __forceinline__ void tdm_load_2d(unsigned lds_addr, const void* gaddr,
                                            unsigned tile0, unsigned tile1,
                                            unsigned td0, unsigned td1,
                                            unsigned long long stride0) {
  unsigned long long ga = (unsigned long long)gaddr;
  v4u g0;
  g0[0] = 1u;                                                  // count=1, user D#
  g0[1] = lds_addr;                                            // LDS byte offset
  g0[2] = (unsigned)ga;                                        // global_addr[31:0]
  g0[3] = ((unsigned)(ga >> 32) & 0x01FFFFFFu) | (2u << 30);   // addr[56:32] | type=2

  v8i g1;
  g1[0] = (int)(1u << 16);                                     // data_size=1 (2B)
  g1[1] = (int)((td0 & 0xFFFFu) << 16);                        // tensor_dim0[15:0]
  g1[2] = (int)((td0 >> 16) | ((td1 & 0xFFFFu) << 16));        // td0[31:16]|td1[15:0]
  g1[3] = (int)((td1 >> 16) | (tile0 << 16));                  // td1[31:16]|tile_dim0
  g1[4] = (int)(tile1 & 0xFFFFu);                              // tile_dim1, tile_dim2=0
  g1[5] = (int)(unsigned)(stride0 & 0xFFFFFFFFull);            // dim0_stride[31:0]
  g1[6] = (int)(unsigned)(stride0 >> 32);                      // dim0_stride[47:32]
  g1[7] = 0;                                                   // dim1_stride (2-D)

  v4i zero4 = {0, 0, 0, 0};
#if __has_include(<hip/amd_detail/amd_gfx1250_TDM.h>)
  v8i zero8 = {0, 0, 0, 0, 0, 0, 0, 0};
  __builtin_amdgcn_tensor_load_to_lds(g0, g1, zero4, zero4, zero8, 0);
#else
  __builtin_amdgcn_tensor_load_to_lds(g0, g1, zero4, zero4, 0);
#endif
}
#endif

// ---------------- fp32 -> bf16 conversion (4 elems / thread) ----------------
__global__ void cvt_f32_bf16_x4(const float* __restrict__ in,
                                unsigned short* __restrict__ out, int n) {
  int i = (blockIdx.x * blockDim.x + threadIdx.x) * 4;
  if (i + 3 < n) {
    float4 v = *(const float4*)(in + i);
    uint2 r;
    r.x = (unsigned)f32_to_bf16(v.x) | ((unsigned)f32_to_bf16(v.y) << 16);
    r.y = (unsigned)f32_to_bf16(v.z) | ((unsigned)f32_to_bf16(v.w) << 16);
    *(uint2*)(out + i) = r;
  }
}

// ---------------- QKV projection GEMM:  C[8192,1024] = E * W ----------------
// Tile 128(M) x 64(N), K-step 32.  8 waves: wave w owns rows w*16..w*16+15.
// A tile staged by the Tensor Data Mover (wave 0), B tile transposed by ALUs.
// z=0 -> Q (scale 1/32 folded in), z=1 -> K, z=2 -> V stored transposed [B][D][S].
__launch_bounds__(256)
__global__ void qkv_gemm(const unsigned short* __restrict__ E,
                         const unsigned short* __restrict__ Wq,
                         const unsigned short* __restrict__ Wk,
                         const unsigned short* __restrict__ Wv,
                         unsigned short* __restrict__ Qo,
                         unsigned short* __restrict__ Ko,
                         unsigned short* __restrict__ VtO) {
  const int z = blockIdx.z;
  const unsigned short* W   = (z == 0) ? Wq : (z == 1) ? Wk : Wv;
  unsigned short*       Out = (z == 0) ? Qo : (z == 1) ? Ko : VtO;
  const float scale = (z == 0) ? 0.03125f : 1.0f;   // 1/sqrt(1024) on Q only

  __shared__ __align__(16) unsigned short ldsA[128][32];
  __shared__ __align__(16) unsigned short ldsBT[64][32];   // B transposed: [n][k]

  const int tid  = threadIdx.x;
  const int lane = tid & 31;
  const int w    = tid >> 5;
  const int m0   = blockIdx.y * 128;
  const int n0   = blockIdx.x * 64;

  v8f acc[4] = {};

  const int arow = tid >> 1, acol = (tid & 1) * 16;   // A: 128 rows x 32
  const int brow = tid >> 3, bcol = (tid & 7) * 8;    // B: 32 rows x 64

#if HAVE_TDM
  const unsigned ldsA_off = (unsigned)(unsigned long long)(void*)&ldsA[0][0];
#endif

  for (int k0 = 0; k0 < DIM; k0 += 32) {
    __syncthreads();
#if HAVE_TDM
    if (w == 0) {     // TDM DMA: 128x32 bf16 tile -> linear LDS
      tdm_load_2d(ldsA_off, E + (size_t)m0 * DIM + k0,
                  /*tile*/ 32u, 128u, /*tensor*/ 32u, 128u, /*stride*/ DIM);
    }
#else
    { // stage A tile (coalesced 16B loads -> ds_store_b128)
      const unsigned short* src = E + (size_t)(m0 + arow) * DIM + k0 + acol;
      *(uint4*)&ldsA[arow][acol]     = *(const uint4*)(src);
      *(uint4*)&ldsA[arow][acol + 8] = *(const uint4*)(src + 8);
    }
#endif
    { // stage B tile transposed
      uint4 v = *(const uint4*)(W + (size_t)(k0 + brow) * DIM + n0 + bcol);
      unsigned short tmp[8];
      *(uint4*)tmp = v;
      #pragma unroll
      for (int j = 0; j < 8; ++j) ldsBT[bcol + j][brow] = tmp[j];
    }
#if HAVE_TDM
    if (w == 0) __builtin_amdgcn_s_wait_tensorcnt(0);
#endif
    __syncthreads();

    v16bf a  = load_frag_a(&ldsA[w * 16][0], 32, lane);
    v16bf b0 = load_frag_b(&ldsBT[0][0],  32, lane);
    v16bf b1 = load_frag_b(&ldsBT[16][0], 32, lane);
    v16bf b2 = load_frag_b(&ldsBT[32][0], 32, lane);
    v16bf b3 = load_frag_b(&ldsBT[48][0], 32, lane);
    acc[0] = __builtin_amdgcn_wmma_f32_16x16x32_bf16(false, a, false, b0, (short)0, acc[0], false, false);
    acc[1] = __builtin_amdgcn_wmma_f32_16x16x32_bf16(false, a, false, b1, (short)0, acc[1], false, false);
    acc[2] = __builtin_amdgcn_wmma_f32_16x16x32_bf16(false, a, false, b2, (short)0, acc[2], false, false);
    acc[3] = __builtin_amdgcn_wmma_f32_16x16x32_bf16(false, a, false, b3, (short)0, acc[3], false, false);
  }

  const int half = lane >> 4, ln = lane & 15;
  #pragma unroll
  for (int nt = 0; nt < 4; ++nt) {
    #pragma unroll
    for (int j = 0; j < 8; ++j) {
      int mg = m0 + w * 16 + j + half * 8;            // C layout: vgpr j -> M=j(+8)
      int ng = n0 + nt * 16 + ln;
      unsigned short bv = f32_to_bf16(acc[nt][j] * scale);
      if (z != 2) {
        Out[(size_t)mg * DIM + ng] = bv;
      } else {                                        // V^T: [b][d][s]
        int bb = mg >> 12, ss = mg & 4095;
        Out[((size_t)bb * DIM + ng) * SEQ + ss] = bv;
      }
    }
  }
}

// ---------------- flash attention ----------------
// 256 thr = 8 waves.  Workgroup: 32 queries.  wave -> (qw = w&1 : 16-query
// subtile, kd = w>>1 : 16-key score subtile AND 256-wide d-chunk for PV).
__launch_bounds__(256)
__global__ void flash_attn(const unsigned short* __restrict__ Q,
                           const unsigned short* __restrict__ K,
                           const unsigned short* __restrict__ Vt,
                           const float* __restrict__ mask,
                           float* __restrict__ Out) {
  const int tid  = threadIdx.x;
  const int lane = tid & 31;
  const int w    = tid >> 5;
  const int qw   = w & 1;
  const int kd   = w >> 1;                 // 0..3
  const int b    = blockIdx.y;
  const int qbase = blockIdx.x * 32 + qw * 16;
  const int half = lane >> 4, ln = lane & 15;

  __shared__ __align__(16) unsigned short sp[2][16][64];  // P tile, bf16
  __shared__ float stmax[2][4][16];
  __shared__ float sm[2][16], sl[2][16], scf[2][16], smn[2][16];

  if (tid < 32) {
    sm[tid >> 4][tid & 15] = -__builtin_inff();
    sl[tid >> 4][tid & 15] = 0.0f;
  }
  __syncthreads();

  v8f acc[16] = {};                                       // 16 x 256 fp32 O-chunk

  const unsigned short* Qrow = Q + ((size_t)b * SEQ + qbase) * DIM;

  for (int kt = 0; kt < SEQ / 64; ++kt) {
    const int k0 = kt * 64;
    const int kb16 = k0 + kd * 16;

    // prefetch next key tile's K rows into cache (global_prefetch_b8)
    if (kt + 1 < SEQ / 64) {
      const unsigned short* nk = K + ((size_t)b * SEQ + kb16 + 64 + ln) * DIM;
      __builtin_prefetch(nk, 0, 1);
      __builtin_prefetch(nk + DIM / 2, 0, 1);
    }

    // ----- S = (Q*scale) . K^T  : 16x16 tile, K-dim = 1024 -----
    v8f s = {};
    const unsigned short* Krow = K + ((size_t)b * SEQ + kb16) * DIM;
    #pragma unroll 4
    for (int d0 = 0; d0 < DIM; d0 += 32) {
      v16bf a  = load_frag_a(Qrow + d0, DIM, lane);
      v16bf bb = load_frag_b(Krow + d0, DIM, lane);   // K rows ARE B columns
      s = __builtin_amdgcn_wmma_f32_16x16x32_bf16(
          false, a, false, bb, (short)0, s, false, false);
    }

    // ----- faithful mask: where(mask==0,-inf,mask).T added to scores -----
    #pragma unroll
    for (int i = 0; i < 8; ++i) {
      int qg = qbase + i + half * 8;
      int kg = kb16 + ln;
      float mv = mask[(size_t)kg * SEQ + qg];
      s[i] += (mv == 0.0f) ? -__builtin_inff() : mv;
    }

    // ----- per-row max inside 16x16 tile (reduce across 16-lane halves) -----
    float rmax[8];
    #pragma unroll
    for (int i = 0; i < 8; ++i) {
      float r = s[i];
      r = fmaxf(r, __shfl_xor(r, 1, 32));
      r = fmaxf(r, __shfl_xor(r, 2, 32));
      r = fmaxf(r, __shfl_xor(r, 4, 32));
      r = fmaxf(r, __shfl_xor(r, 8, 32));
      rmax[i] = r;
    }
    if (ln == 0) {
      #pragma unroll
      for (int i = 0; i < 8; ++i) stmax[qw][kd][i + half * 8] = rmax[i];
    }
    __syncthreads();                                   // B1

    if (kd == 0 && lane < 16) {                        // online-softmax state
      int q = lane;
      float mo = sm[qw][q];
      float t  = fmaxf(fmaxf(stmax[qw][0][q], stmax[qw][1][q]),
                       fmaxf(stmax[qw][2][q], stmax[qw][3][q]));
      float mn = fmaxf(mo, t);
      float cf = __expf(mo - mn);                      // first iter: exp(-inf)=0
      smn[qw][q] = mn;  scf[qw][q] = cf;
      sm[qw][q]  = mn;  sl[qw][q] *= cf;
    }
    __syncthreads();                                   // B2

    // ----- P = exp(S - mnew) -> LDS (bf16), l += rowsum -----
    #pragma unroll
    for (int i = 0; i < 8; ++i) {
      int qr = i + half * 8;
      float p = __expf(s[i] - smn[qw][qr]);
      sp[qw][qr][kd * 16 + ln] = f32_to_bf16(p);
      float r = p;
      r += __shfl_xor(r, 1, 32);
      r += __shfl_xor(r, 2, 32);
      r += __shfl_xor(r, 4, 32);
      r += __shfl_xor(r, 8, 32);
      if (ln == 0) atomicAdd(&sl[qw][qr], r);
    }
    // ----- rescale O accumulators by correction factor -----
    float cfa[8];
    #pragma unroll
    for (int j = 0; j < 8; ++j) cfa[j] = scf[qw][j + half * 8];
    #pragma unroll
    for (int nt = 0; nt < 16; ++nt)
      #pragma unroll
      for (int j = 0; j < 8; ++j) acc[nt][j] *= cfa[j];
    __syncthreads();                                   // B3: P complete

    // ----- O += P @ V   (wave's 256-wide d-chunk; V pre-transposed) -----
    const unsigned short* pbase = &sp[qw][0][0];
    v16bf pa0 = load_frag_a(pbase,      64, lane);     // P frags invariant
    v16bf pa1 = load_frag_a(pbase + 32, 64, lane);     // across the 16 d-tiles
    #pragma unroll 4
    for (int nt = 0; nt < 16; ++nt) {
      const int dbase = kd * 256 + nt * 16;
      const unsigned short* vb = Vt + ((size_t)b * DIM + dbase) * SEQ + k0;
      v16bf vb0 = load_frag_b(vb,      SEQ, lane);
      v16bf vb1 = load_frag_b(vb + 32, SEQ, lane);
      acc[nt] = __builtin_amdgcn_wmma_f32_16x16x32_bf16(
          false, pa0, false, vb0, (short)0, acc[nt], false, false);
      acc[nt] = __builtin_amdgcn_wmma_f32_16x16x32_bf16(
          false, pa1, false, vb1, (short)0, acc[nt], false, false);
    }
  }

  __syncthreads();
  float linv[8];
  #pragma unroll
  for (int j = 0; j < 8; ++j) linv[j] = 1.0f / sl[qw][j + half * 8];
  #pragma unroll
  for (int nt = 0; nt < 16; ++nt) {
    #pragma unroll
    for (int j = 0; j < 8; ++j) {
      int qg = qbase + j + half * 8;
      int dg = kd * 256 + nt * 16 + ln;
      Out[((size_t)b * SEQ + qg) * DIM + dg] = acc[nt][j] * linv[j];
    }
  }
}

// ---------------- host ----------------
extern "C" void kernel_launch(void* const* d_in, const int* in_sizes, int n_in,
                              void* d_out, int out_size, void* d_ws, size_t ws_size,
                              hipStream_t stream) {
  (void)in_sizes; (void)n_in; (void)out_size; (void)ws_size;
  const float* emb  = (const float*)d_in[0];
  const float* mask = (const float*)d_in[1];
  const float* wq   = (const float*)d_in[2];
  const float* wk   = (const float*)d_in[3];
  const float* wv   = (const float*)d_in[4];
  float* out = (float*)d_out;

  char* ws = (char*)d_ws;                       // 70 MB used
  unsigned short* Ebf = (unsigned short*)(ws);
  unsigned short* Qbf = (unsigned short*)(ws + (size_t)16 * 1024 * 1024);
  unsigned short* Kbf = (unsigned short*)(ws + (size_t)32 * 1024 * 1024);
  unsigned short* Vtb = (unsigned short*)(ws + (size_t)48 * 1024 * 1024);
  unsigned short* Wqb = (unsigned short*)(ws + (size_t)64 * 1024 * 1024);
  unsigned short* Wkb = (unsigned short*)(ws + (size_t)66 * 1024 * 1024);
  unsigned short* Wvb = (unsigned short*)(ws + (size_t)68 * 1024 * 1024);

  const int nE = BATCH * SEQ * DIM;   // 8,388,608
  const int nW = DIM * DIM;           // 1,048,576
  cvt_f32_bf16_x4<<<dim3(nE / 1024), dim3(256), 0, stream>>>(emb, Ebf, nE);
  cvt_f32_bf16_x4<<<dim3(nW / 1024), dim3(256), 0, stream>>>(wq, Wqb, nW);
  cvt_f32_bf16_x4<<<dim3(nW / 1024), dim3(256), 0, stream>>>(wk, Wkb, nW);
  cvt_f32_bf16_x4<<<dim3(nW / 1024), dim3(256), 0, stream>>>(wv, Wvb, nW);

  qkv_gemm<<<dim3(DIM / 64, (BATCH * SEQ) / 128, 3), dim3(256), 0, stream>>>(
      Ebf, Wqb, Wkb, Wvb, Qbf, Kbf, Vtb);

  flash_attn<<<dim3(SEQ / 32, BATCH), dim3(256), 0, stream>>>(
      Qbf, Kbf, Vtb, mask, out);
}
